// CSWALayer_74594991996965
// MI455X (gfx1250) — matrix operand order repaired and checked by reading, not compiled, not run on hardware
//
#include <hip/hip_runtime.h>
#include <hip/hip_bf16.h>

// CSWALayer forward for MI455X (gfx1250, wave32, WMMA).
// Heavy GEMM/conv work on V_WMMA_F32_16X16X32_F16; LDS double-buffered,
// 32x32 output tile per wave (4 accumulators -> 4 wmma per 8 ds_load_b128).

typedef _Float16 h16;
typedef __attribute__((ext_vector_type(16))) _Float16 v16h;
typedef __attribute__((ext_vector_type(4)))  _Float16 v4h;
typedef __attribute__((ext_vector_type(8)))  float    v8f;
typedef __attribute__((ext_vector_type(4)))  float    v4f;

#define CDIV(a, b) (((a) + (b) - 1) / (b))

__device__ __forceinline__ v8f wmma16(v16h a, v16h b, v8f c) {
  // D = A(16x32 f16) x B(32x16 f16) + C(16x16 f32)
  return __builtin_amdgcn_wmma_f32_16x16x32_f16(false, a, false, b, (short)0, c,
                                                false, false);
}

// A fragment (16x32 f16) from LDS rows [m][k], ld in halves (multiple of 8).
// lane<16: halves K0..7 @+0, K16..23 @+32B ; lane>=16: +8 halves (K8..15,K24..31)
__device__ __forceinline__ v16h frag_a(const h16* base, int ld) {
  int lane = threadIdx.x & 31;
  const h16* p = base + (lane & 15) * ld + ((lane >> 4) << 3);
  union { v16h h; v4f f[2]; } u;
  u.f[0] = *(const v4f*)(p);
  u.f[1] = *(const v4f*)(p + 16);
  return u.h;
}

// B fragment (32x16 f16) from LDS rows [n][k] (column-per-lane):
// lane n<16 reads K0..7, K16..23; lane>=16 reads K8..15, K24..31 (+8 halves).
__device__ __forceinline__ v16h frag_b(const h16* base, int ld) {
  int lane = threadIdx.x & 31;
  const h16* p = base + (lane & 15) * ld + ((lane >> 4) << 3);
  union { v16h h; v4f f[2]; } u;
  u.f[0] = *(const v4f*)(p);
  u.f[1] = *(const v4f*)(p + 16);
  return u.h;
}

// ---------------- elementwise / transform kernels ----------------

// OIHW f32 -> [O][ky][kx][I] f16 with BN scale folded into weights.
__global__ void k_wt3(const float* __restrict__ w, const float* __restrict__ s,
                      h16* __restrict__ dst, int O, int Cin) {
  int idx = blockIdx.x * 256 + threadIdx.x;
  int total = O * 9 * Cin;
  if (idx >= total) return;
  int ic = idx % Cin;
  int tap = (idx / Cin) % 9;
  int oc = idx / (9 * Cin);
  dst[idx] = (h16)(w[(size_t)(oc * Cin + ic) * 9 + tap] * s[oc]);
}

__global__ void k_wt1(const float* __restrict__ w, h16* __restrict__ dst, int n) {
  int idx = blockIdx.x * 256 + threadIdx.x;
  if (idx < n) dst[idx] = (h16)w[idx];
}

// NCHW f32 -> NHWC f16
__global__ void k_cvt(const float* __restrict__ src, h16* __restrict__ dst,
                      int C, int HW, int total) {
  int idx = blockIdx.x * 256 + threadIdx.x;
  if (idx >= total) return;
  int c = idx % C;
  int pix = (idx / C) % HW;
  int b = idx / (C * HW);
  dst[idx] = (h16)src[((size_t)b * C + c) * HW + pix];
}

// p1g[b][d][s] = sum of the 4 p1 columns that the tile(1,1,2,2) maps onto s.
// d = c*16 + kh*4 + kw over f11 (NHWC,128ch,80x80); s in [0,128) (>=100 -> 0).
__global__ void k_p1gather(const h16* __restrict__ f11, h16* __restrict__ p1g) {
  int idx = blockIdx.x * 256 + threadIdx.x;
  if (idx >= 32 * 2048 * 128) return;
  int s = idx & 127;
  int d = (idx >> 7) & 2047;
  int b = idx >> 18;
  float v = 0.f;
  if (s < 100) {
    int c = d >> 4, t = d & 15;
    int kh = t >> 2, kw = t & 3;
    int rh = s / 10, rw = s % 10;
    const h16* fb = f11 + (size_t)b * 6400 * 128;
    for (int dh = 0; dh < 2; ++dh)
      for (int dw = 0; dw < 2; ++dw) {
        int y = (rh + dh * 10) * 4 + kh;
        int x = (rw + dw * 10) * 4 + kw;
        v += (float)fb[(size_t)(y * 80 + x) * 128 + c];
      }
  }
  p1g[idx] = (h16)v;
}

// t[b][q][d] -> tp[(b*400 + y*20 + x)][c]  (inverse of patch extraction)
__global__ void k_repack(const h16* __restrict__ t, h16* __restrict__ tp) {
  int idx = blockIdx.x * 256 + threadIdx.x;
  if (idx >= 12800 * 128) return;
  int c = idx & 127;
  int pix = idx >> 7;
  int x = pix % 20;
  int y = (pix / 20) % 20;
  int b = pix / 400;
  int q = (y >> 2) * 5 + (x >> 2);
  int d = c * 16 + (y & 3) * 4 + (x & 3);
  tp[idx] = t[((size_t)b * 32 + q) * 2048 + d];
}

// out[b][ch][y][x] = dot(h[pixel][512], wd2[ch][512])   (N=2 -> plain VALU)
__global__ void k_final(const h16* __restrict__ h, const float* __restrict__ wd2,
                        float* __restrict__ out) {
  int idx = blockIdx.x * 256 + threadIdx.x;
  if (idx >= 32 * 2 * 400) return;
  int pix = idx % 400;
  int ch = (idx / 400) & 1;
  int b = idx / 800;
  const h16* hr = h + ((size_t)b * 400 + pix) * 512;
  const float* wr = wd2 + ch * 512;
  float acc = 0.f;
#pragma unroll 8
  for (int k = 0; k < 512; ++k) acc += (float)hr[k] * wr[k];
  out[idx] = acc;
}

// ---------------- WMMA implicit-GEMM 3x3 conv + BN + SiLU ----------------
// in: NHWC f16, wt: [O][3][3][I] f16 (scale folded), bias: f32[O], out NHWC f16.
// grid: (ceil(HW/64), Cout/128, B), block 256 = 8 waves in 2(M) x 4(N);
// each wave owns a 32x32 output tile (4 f32 accumulators).
// LDS is double-buffered: one barrier per K-step; next K-step global loads
// are issued before the current step's WMMAs.
__global__ __launch_bounds__(256) void conv3x3_silu_wmma(
    const h16* __restrict__ in, const h16* __restrict__ wt,
    const float* __restrict__ bias, h16* __restrict__ out,
    int H, int W, int Cin, int Cout) {
  __shared__ h16 shA[2][64 * 40];
  __shared__ h16 shB[2][128 * 40];
  const int HW = H * W;
  const int b = blockIdx.z;
  const int pix0 = blockIdx.x * 64;
  const int oc0 = blockIdx.y * 128;
  const int tid = threadIdx.x;
  const int lane = tid & 31;
  const int wid = tid >> 5;
  const int wm = wid >> 2;  // 0..1 : 32-pixel sub-tile
  const int wn = wid & 3;   // 0..3 : 32-cout sub-tile

  const int arow = tid >> 2, aseg = (tid & 3) << 3;  // A: 64 rows x 4 segs
  const int brow = tid >> 1, bseg = (tid & 1) << 4;  // B: 128 rows x 2x16h

  const int apix = pix0 + arow;
  const bool pixok = apix < HW;
  const int ay = pixok ? (apix / W) : 0;
  const int ax = pixok ? (apix % W) : 0;

  v8f acc00 = {}, acc01 = {}, acc10 = {}, acc11 = {};
  const int kcN = Cin >> 5;
  const h16* inB = in + (size_t)b * HW * Cin;
  const v4f vzero = {0.f, 0.f, 0.f, 0.f};

  int p = 0;
  for (int tap = 0; tap < 9; ++tap) {
    const int ky = tap / 3, kx = tap % 3;
    const int sy = ay + ky - 1, sx = ax + kx - 1;
    const bool tapok = pixok && (unsigned)sy < (unsigned)H && (unsigned)sx < (unsigned)W;
    const h16* arowp = inB + (size_t)(sy * W + sx) * Cin;
    const h16* browp = wt + (size_t)((oc0 + brow) * 9 + tap) * Cin;
    if (tap < 8)
      __builtin_prefetch(wt + (size_t)((oc0 + brow) * 9 + tap + 1) * Cin + bseg, 0, 0);

    // prologue: stage kc=0 into buffer p
    v4f av = vzero;
    if (tapok) av = *(const v4f*)(arowp + aseg);
    v4f bv0 = *(const v4f*)(browp + bseg);
    v4f bv1 = *(const v4f*)(browp + bseg + 8);
    __syncthreads();  // protect buffer p from previous tap's readers
    *(v4f*)(&shA[p][arow * 40 + aseg]) = av;
    *(v4f*)(&shB[p][brow * 40 + bseg]) = bv0;
    *(v4f*)(&shB[p][brow * 40 + bseg + 8]) = bv1;

    for (int kc = 0; kc < kcN; ++kc) {
      __syncthreads();
      const bool more = (kc + 1) < kcN;
      if (more) {  // issue next K-step's global loads early
        const int ko = (kc + 1) << 5;
        av = vzero;
        if (tapok) av = *(const v4f*)(arowp + ko + aseg);
        bv0 = *(const v4f*)(browp + ko + bseg);
        bv1 = *(const v4f*)(browp + ko + bseg + 8);
      }
      v16h fa0 = frag_a(&shA[p][(wm * 32) * 40], 40);
      v16h fa1 = frag_a(&shA[p][(wm * 32 + 16) * 40], 40);
      v16h fb0 = frag_b(&shB[p][(wn * 32) * 40], 40);
      v16h fb1 = frag_b(&shB[p][(wn * 32 + 16) * 40], 40);
      acc00 = wmma16(fa0, fb0, acc00);
      acc01 = wmma16(fa0, fb1, acc01);
      acc10 = wmma16(fa1, fb0, acc10);
      acc11 = wmma16(fa1, fb1, acc11);
      if (more) {
        p ^= 1;
        *(v4f*)(&shA[p][arow * 40 + aseg]) = av;
        *(v4f*)(&shB[p][brow * 40 + bseg]) = bv0;
        *(v4f*)(&shB[p][brow * 40 + bseg + 8]) = bv1;
      }
    }
  }

  const int ml = lane & 15, mh = (lane >> 4) << 3;
#pragma unroll
  for (int mi = 0; mi < 2; ++mi) {
#pragma unroll
    for (int sub = 0; sub < 2; ++sub) {
      v8f a = mi == 0 ? (sub == 0 ? acc00 : acc01) : (sub == 0 ? acc10 : acc11);
      int oc = oc0 + wn * 32 + sub * 16 + ml;
      float bvs = bias[oc];
#pragma unroll
      for (int i = 0; i < 8; ++i) {
        int pix = pix0 + wm * 32 + mi * 16 + i + mh;
        if (pix < HW) {
          float v = a[i] + bvs;
          v = v / (1.0f + __expf(-v));  // SiLU
          out[((size_t)b * HW + pix) * Cout + oc] = (h16)v;
        }
      }
    }
  }
}

// ---------------- attention scores + softmax (per-batch block) ----------------
// S = (p3^T . p2)/sqrt(2048), tiled-400 softmax folded to exp/(4*sum100).
// probs[b][32][128] f16 (q>=25 or k>=100 -> 0).
__global__ __launch_bounds__(256) void attn_scores(
    const h16* __restrict__ f33, const h16* __restrict__ f22,
    h16* __restrict__ probs) {
  __shared__ h16 shA[32 * 40];
  __shared__ h16 shB[128 * 40];
  __shared__ float shS[32 * 128];
  const int b = blockIdx.x;
  const int tid = threadIdx.x;
  const int wid = tid >> 5, lane = tid & 31;
  v8f acc0 = {}, acc1 = {};
  const h16* f33b = f33 + (size_t)b * 400 * 128;
  const h16* f22b = f22 + (size_t)b * 1600 * 128;
  const int am = tid >> 3, ak = (tid & 7) << 2;   // A: 32x32 tile
  const int ahb = am / 5, awb = am % 5;
  const int bn = tid >> 1, bk = (tid & 1) << 4;   // B: 128x32 tile
  const int bhb = bn / 10, bwb = bn % 10;

  for (int kc = 0; kc < 64; ++kc) {
    const int K0 = kc << 5;
    h16 abuf[4];
#pragma unroll
    for (int j = 0; j < 4; ++j) {
      int k = K0 + ak + j;
      int c = k >> 4, t = k & 15;
      int y = ahb * 4 + (t >> 2), x = awb * 4 + (t & 3);
      abuf[j] = (am < 25) ? f33b[(size_t)(y * 20 + x) * 128 + c] : (h16)0.f;
    }
    h16 bbuf[16];
#pragma unroll
    for (int j = 0; j < 16; ++j) {
      int k = K0 + bk + j;
      int c = k >> 4, t = k & 15;
      int y = bhb * 4 + (t >> 2), x = bwb * 4 + (t & 3);
      bbuf[j] = (bn < 100) ? f22b[(size_t)(y * 40 + x) * 128 + c] : (h16)0.f;
    }
    __syncthreads();
#pragma unroll
    for (int j = 0; j < 4; ++j) shA[am * 40 + ak + j] = abuf[j];
#pragma unroll
    for (int j = 0; j < 16; ++j) shB[bn * 40 + bk + j] = bbuf[j];
    __syncthreads();
    v16h fa0 = frag_a(shA, 40);
    v16h fa1 = frag_a(shA + 16 * 40, 40);
    v16h fb  = frag_b(shB + wid * 16 * 40, 40);
    acc0 = wmma16(fa0, fb, acc0);
    acc1 = wmma16(fa1, fb, acc1);
  }
  const float scale = 0.02209708691207961f;  // 1/sqrt(2048)
  const int ml = lane & 15, mh = (lane >> 4) << 3;
#pragma unroll
  for (int i = 0; i < 8; ++i) {
    shS[(i + mh) * 128 + wid * 16 + ml] = acc0[i] * scale;
    shS[(i + mh + 16) * 128 + wid * 16 + ml] = acc1[i] * scale;
  }
  __syncthreads();
  if (tid < 32) {
    int q = tid;
    h16* prow = probs + ((size_t)b * 32 + q) * 128;
    if (q < 25) {
      float mx = -1e30f;
      for (int k = 0; k < 100; ++k) mx = fmaxf(mx, shS[q * 128 + k]);
      float s = 0.f;
      for (int k = 0; k < 100; ++k) s += __expf(shS[q * 128 + k] - mx);
      float inv = 1.0f / (4.0f * s);
      for (int k = 0; k < 128; ++k)
        prow[k] = (k < 100) ? (h16)(__expf(shS[q * 128 + k] - mx) * inv) : (h16)0.f;
    } else {
      for (int k = 0; k < 128; ++k) prow[k] = (h16)0.f;
    }
  }
}

// t[b][q][d] = probs . p1g^T  (M=32, K=128 padded, N=2048) -- grid (16, B)
__global__ __launch_bounds__(256) void attn_out_wmma(
    const h16* __restrict__ probs, const h16* __restrict__ p1g,
    h16* __restrict__ t) {
  __shared__ h16 shP[32 * 40];
  __shared__ h16 shG[128 * 40];
  const int b = blockIdx.y;
  const int n0 = blockIdx.x * 128;
  const int tid = threadIdx.x;
  const int wid = tid >> 5, lane = tid & 31;
  v8f acc0 = {}, acc1 = {};
  const int am = tid >> 3, ak = (tid & 7) << 2;
  const int bn = tid >> 1, bk = (tid & 1) << 4;
  for (int kc = 0; kc < 4; ++kc) {
    const int K0 = kc << 5;
    v4h pv = *(const v4h*)(probs + ((size_t)b * 32 + am) * 128 + K0 + ak);
    const h16* gp = p1g + ((size_t)b * 2048 + n0 + bn) * 128 + K0 + bk;
    v4f g0 = *(const v4f*)(gp);
    v4f g1 = *(const v4f*)(gp + 8);
    __syncthreads();
    *(v4h*)(&shP[am * 40 + ak]) = pv;
    *(v4f*)(&shG[bn * 40 + bk]) = g0;
    *(v4f*)(&shG[bn * 40 + bk + 8]) = g1;
    __syncthreads();
    v16h fa0 = frag_a(shP, 40);
    v16h fa1 = frag_a(shP + 16 * 40, 40);
    v16h fb  = frag_b(shG + wid * 16 * 40, 40);
    acc0 = wmma16(fa0, fb, acc0);
    acc1 = wmma16(fa1, fb, acc1);
  }
  const int ml = lane & 15, mh = (lane >> 4) << 3;
#pragma unroll
  for (int i = 0; i < 8; ++i) {
    t[((size_t)b * 32 + i + mh) * 2048 + n0 + wid * 16 + ml] = (h16)acc0[i];
    t[((size_t)b * 32 + i + mh + 16) * 2048 + n0 + wid * 16 + ml] = (h16)acc1[i];
  }
}

// h = relu(tp . wd1t^T):  A[M][K] f16, Bw[N][K] f16, Out[M][N] f16.
// grid (M/64, N/128), 32x32 wave tiles, double-buffered LDS.
__global__ __launch_bounds__(256) void gemm_relu_wmma(
    const h16* __restrict__ A, const h16* __restrict__ Bw,
    h16* __restrict__ Out, int M, int N, int K) {
  __shared__ h16 shA[2][64 * 40];
  __shared__ h16 shB[2][128 * 40];
  const int tid = threadIdx.x;
  const int wid = tid >> 5, lane = tid & 31;
  const int wm = wid >> 2, wn = wid & 3;
  const int m0 = blockIdx.x * 64, n0 = blockIdx.y * 128;
  const int arow = tid >> 2, aseg = (tid & 3) << 3;
  const int brow = tid >> 1, bseg = (tid & 1) << 4;
  v8f acc00 = {}, acc01 = {}, acc10 = {}, acc11 = {};
  const int kcN = K >> 5;
  const h16* arowp = A + (size_t)(m0 + arow) * K;
  const h16* browp = Bw + (size_t)(n0 + brow) * K;

  int p = 0;
  v4f av = *(const v4f*)(arowp + aseg);
  v4f bv0 = *(const v4f*)(browp + bseg);
  v4f bv1 = *(const v4f*)(browp + bseg + 8);
  *(v4f*)(&shA[p][arow * 40 + aseg]) = av;
  *(v4f*)(&shB[p][brow * 40 + bseg]) = bv0;
  *(v4f*)(&shB[p][brow * 40 + bseg + 8]) = bv1;

  for (int kc = 0; kc < kcN; ++kc) {
    __syncthreads();
    const bool more = (kc + 1) < kcN;
    if (more) {
      const int ko = (kc + 1) << 5;
      av = *(const v4f*)(arowp + ko + aseg);
      bv0 = *(const v4f*)(browp + ko + bseg);
      bv1 = *(const v4f*)(browp + ko + bseg + 8);
    }
    v16h fa0 = frag_a(&shA[p][(wm * 32) * 40], 40);
    v16h fa1 = frag_a(&shA[p][(wm * 32 + 16) * 40], 40);
    v16h fb0 = frag_b(&shB[p][(wn * 32) * 40], 40);
    v16h fb1 = frag_b(&shB[p][(wn * 32 + 16) * 40], 40);
    acc00 = wmma16(fa0, fb0, acc00);
    acc01 = wmma16(fa0, fb1, acc01);
    acc10 = wmma16(fa1, fb0, acc10);
    acc11 = wmma16(fa1, fb1, acc11);
    if (more) {
      p ^= 1;
      *(v4f*)(&shA[p][arow * 40 + aseg]) = av;
      *(v4f*)(&shB[p][brow * 40 + bseg]) = bv0;
      *(v4f*)(&shB[p][brow * 40 + bseg + 8]) = bv1;
    }
  }
  const int ml = lane & 15, mh = (lane >> 4) << 3;
#pragma unroll
  for (int mi = 0; mi < 2; ++mi) {
#pragma unroll
    for (int sub = 0; sub < 2; ++sub) {
      v8f a = mi == 0 ? (sub == 0 ? acc00 : acc01) : (sub == 0 ? acc10 : acc11);
      int n = n0 + wn * 32 + sub * 16 + ml;
#pragma unroll
      for (int i = 0; i < 8; ++i) {
        int m = m0 + wm * 32 + mi * 16 + i + mh;
        float v = a[i];
        Out[(size_t)m * N + n] = (h16)(v > 0.f ? v : 0.f);
      }
    }
  }
}

// ---------------- host launcher ----------------
extern "C" void kernel_launch(void* const* d_in, const int* in_sizes, int n_in,
                              void* d_out, int out_size, void* d_ws, size_t ws_size,
                              hipStream_t stream) {
  const float* feature1 = (const float*)d_in[0];
  const float* feature2 = (const float*)d_in[1];
  const float* feature3 = (const float*)d_in[2];
  const float* w1  = (const float*)d_in[3];
  const float* s1  = (const float*)d_in[4];
  const float* b1  = (const float*)d_in[5];
  const float* w2a = (const float*)d_in[6];
  const float* s2a = (const float*)d_in[7];
  const float* b2a = (const float*)d_in[8];
  const float* w2b = (const float*)d_in[9];
  const float* s2b = (const float*)d_in[10];
  const float* b2b = (const float*)d_in[11];
  const float* w3a = (const float*)d_in[12];
  const float* s3a = (const float*)d_in[13];
  const float* b3a = (const float*)d_in[14];
  const float* w3b = (const float*)d_in[15];
  const float* s3b = (const float*)d_in[16];
  const float* b3b = (const float*)d_in[17];
  const float* w3c = (const float*)d_in[18];
  const float* s3c = (const float*)d_in[19];
  const float* b3c = (const float*)d_in[20];
  const float* wd1 = (const float*)d_in[21];
  const float* wd2 = (const float*)d_in[22];
  float* out = (float*)d_out;

  char* ws = (char*)d_ws;
  const size_t MiB = 1ull << 20;
  // weights arena (<10 MiB)
  h16* wt1t = (h16*)(ws + 0);
  h16* wt2a = (h16*)(ws + 294912);
  h16* wt2b = (h16*)(ws + 1474560);
  h16* wt3a = (h16*)(ws + 2064384);
  h16* wt3b = (h16*)(ws + 6782976);
  h16* wt3c = (h16*)(ws + 9142272);
  h16* wd1t = (h16*)(ws + 9732096);
  h16* probs = (h16*)(ws + 9863168);
  // activation arena (buffers reused once dead)
  h16* f1h  = (h16*)(ws + 10 * MiB);        // 50 MiB
  h16* f11  = (h16*)(ws + 60 * MiB);        // 50 MiB
  h16* f2h  = (h16*)(ws + 10 * MiB);        // 25 MiB (after f11)
  h16* f22a = (h16*)(ws + 35 * MiB);        // 25 MiB
  h16* f22  = (h16*)(ws + 10 * MiB);        // 12.5 MiB (after f2h)
  h16* f3h  = (h16*)(ws + 23592960);        // 12.5 MiB
  h16* f33a = (h16*)(ws + 35 * MiB);        // 12.5 MiB (after f22a)
  h16* f33b = (h16*)(ws + 49807360);        // 6.25 MiB
  h16* f33  = (h16*)(ws + 56360960);        // 3.125 MiB
  h16* p1g  = (h16*)(ws + 110 * MiB);       // 16 MiB (f11 still live)
  h16* t    = (h16*)(ws + 60 * MiB);        // 4 MiB (after f11/p1gather)
  h16* tp   = (h16*)(ws + 64 * MiB);        // 3.125 MiB
  h16* hbuf = (h16*)(ws + 68 * MiB);        // 12.5 MiB
  (void)ws_size; (void)in_sizes; (void)n_in; (void)out_size;

  // --- weight transforms (BN scale folded) ---
  k_wt3<<<CDIV(128 * 9 * 128, 256), 256, 0, stream>>>(w1, s1, wt1t, 128, 128);
  k_wt3<<<CDIV(256 * 9 * 256, 256), 256, 0, stream>>>(w2a, s2a, wt2a, 256, 256);
  k_wt3<<<CDIV(128 * 9 * 256, 256), 256, 0, stream>>>(w2b, s2b, wt2b, 128, 256);
  k_wt3<<<CDIV(512 * 9 * 512, 256), 256, 0, stream>>>(w3a, s3a, wt3a, 512, 512);
  k_wt3<<<CDIV(256 * 9 * 512, 256), 256, 0, stream>>>(w3b, s3b, wt3b, 256, 512);
  k_wt3<<<CDIV(128 * 9 * 256, 256), 256, 0, stream>>>(w3c, s3c, wt3c, 128, 256);
  k_wt1<<<CDIV(65536, 256), 256, 0, stream>>>(wd1, wd1t, 65536);

  // --- feature1 chain ---
  k_cvt<<<CDIV(32 * 128 * 6400, 256), 256, 0, stream>>>(feature1, f1h, 128, 6400,
                                                        32 * 128 * 6400);
  conv3x3_silu_wmma<<<dim3(100, 1, 32), 256, 0, stream>>>(f1h, wt1t, b1, f11,
                                                          80, 80, 128, 128);
  k_p1gather<<<CDIV(32 * 2048 * 128, 256), 256, 0, stream>>>(f11, p1g);

  // --- feature2 chain ---
  k_cvt<<<CDIV(32 * 256 * 1600, 256), 256, 0, stream>>>(feature2, f2h, 256, 1600,
                                                        32 * 256 * 1600);
  conv3x3_silu_wmma<<<dim3(25, 2, 32), 256, 0, stream>>>(f2h, wt2a, b2a, f22a,
                                                         40, 40, 256, 256);
  conv3x3_silu_wmma<<<dim3(25, 1, 32), 256, 0, stream>>>(f22a, wt2b, b2b, f22,
                                                         40, 40, 256, 128);

  // --- feature3 chain ---
  k_cvt<<<CDIV(32 * 512 * 400, 256), 256, 0, stream>>>(feature3, f3h, 512, 400,
                                                       32 * 512 * 400);
  conv3x3_silu_wmma<<<dim3(7, 4, 32), 256, 0, stream>>>(f3h, wt3a, b3a, f33a,
                                                        20, 20, 512, 512);
  conv3x3_silu_wmma<<<dim3(7, 2, 32), 256, 0, stream>>>(f33a, wt3b, b3b, f33b,
                                                        20, 20, 512, 256);
  conv3x3_silu_wmma<<<dim3(7, 1, 32), 256, 0, stream>>>(f33b, wt3c, b3c, f33,
                                                        20, 20, 256, 128);

  // --- attention ---
  attn_scores<<<32, 256, 0, stream>>>(f33, f22, probs);
  attn_out_wmma<<<dim3(16, 32), 256, 0, stream>>>(probs, p1g, t);
  k_repack<<<CDIV(12800 * 128, 256), 256, 0, stream>>>(t, tp);

  // --- _ImageDA head ---
  gemm_relu_wmma<<<dim3(200, 4), 256, 0, stream>>>(tp, wd1t, hbuf, 12800, 512, 128);
  k_final<<<CDIV(25600, 256), 256, 0, stream>>>(hbuf, wd2, out);
}
